// TVMShieldedAttentionWithRPE_39745627357527
// MI455X (gfx1250) — matrix operand order repaired and checked by reading, not compiled, not run on hardware
//
#include <hip/hip_runtime.h>
#include <hip/hip_bf16.h>

typedef __attribute__((ext_vector_type(16))) __bf16 v16bf;
typedef __attribute__((ext_vector_type(8)))  float  v8f;

#define DMODEL  512
#define NSEQ    1024
#define NBATCH  8
#define NHEAD   8
#define DKH     64
#define MROWS   8192          // NBATCH*NSEQ
#define NSTAT   128
#define LDB     72            // bf16 LDS row pitch (144B: conflict-free, 16B-aligned)

// ---------------- CDNA5 async global->LDS (ASYNCcnt) -------------------------
__device__ __forceinline__ uint32_t lds_off(const void* p) {
  return (uint32_t)(uintptr_t)p;               // LDS flat addr low 32 bits = offset
}
__device__ __forceinline__ void async_ld_b128(uint64_t sbase, uint32_t voff,
                                              uint32_t loff) {
  asm volatile("global_load_async_to_lds_b128 %0, %1, %2"
               :: "v"(loff), "v"(voff), "s"(sbase) : "memory");
}
__device__ __forceinline__ void wait_async0() {
  asm volatile("s_wait_asynccnt 0" ::: "memory");
}

// ---------------- WMMA bf16 16x16x32 fragment builders (wave32, CDNA5) -------
// Layouts per CDNA5 ISA 7.12.2.
__device__ __forceinline__ v8f wmma_bf16(v16bf a, v16bf b, v8f c) {
  return __builtin_amdgcn_wmma_f32_16x16x32_bf16(false, a, false, b, (short)0, c,
                                                 false, false);
}

__device__ __forceinline__ v16bf a_frag_bf(const __bf16* lds, int ld, int row0, int k0) {
  int lane = threadIdx.x & 31;
  int m  = row0 + (lane & 15);
  int kb = (lane >> 4) * 8;
  v16bf a;
#pragma unroll
  for (int e = 0; e < 16; ++e) {
    int k = k0 + kb + e + ((e >> 3) << 3);     // {0..7,16..23} (+8 for hi lanes)
    a[e] = lds[m * ld + k];
  }
  return a;
}

__device__ __forceinline__ v16bf a_frag_f32(const float* lds, int ld, int row0, int k0) {
  int lane = threadIdx.x & 31;
  int m  = row0 + (lane & 15);
  int kb = (lane >> 4) * 8;
  v16bf a;
#pragma unroll
  for (int e = 0; e < 16; ++e) {
    int k = k0 + kb + e + ((e >> 3) << 3);
    a[e] = (__bf16)lds[m * ld + k];
  }
  return a;
}

// B fragment, lds holds [n][k]  (B[k][n] = lds[n*ld+k])
__device__ __forceinline__ v16bf b_frag_nk_bf(const __bf16* lds, int ld, int n0, int k0) {
  int lane = threadIdx.x & 31;
  int n  = n0 + (lane & 15);
  int kb = (lane >> 4) * 8;
  v16bf b;
#pragma unroll
  for (int e = 0; e < 16; ++e) b[e] = lds[n * ld + k0 + kb + e + ((e >> 3) << 3)];
  return b;
}

// B fragment, lds holds [k][n]  (B[k][n] = lds[k*ld+n])
__device__ __forceinline__ v16bf b_frag_kn_bf(const __bf16* lds, int ld, int n0, int k0) {
  int lane = threadIdx.x & 31;
  int n  = n0 + (lane & 15);
  int kb = (lane >> 4) * 8;
  v16bf b;
#pragma unroll
  for (int e = 0; e < 16; ++e) b[e] = lds[(k0 + kb + e + ((e >> 3) << 3)) * ld + n];
  return b;
}

// ---------------- Kernel 0: one-shot f32 -> bf16 cast ------------------------
__global__ __launch_bounds__(256)
void cvt_bf16_kernel(const float* __restrict__ in, __bf16* __restrict__ out, int n) {
  int i = (blockIdx.x * 256 + threadIdx.x) * 4;
  if (i < n) {
    float4 v = *(const float4*)&in[i];
    union { __bf16 b[4]; uint2 u; } p;
    p.b[0] = (__bf16)v.x; p.b[1] = (__bf16)v.y;
    p.b[2] = (__bf16)v.z; p.b[3] = (__bf16)v.w;
    *(uint2*)&out[i] = p.u;
  }
}

// ---------------- bf16 GEMM body: 128x64 tile, async double-buffered ---------
// O[m,n] = sum_k A[m,k] * W[n,k]   (A: MROWSx512 bf16, W: 512x512 bf16)
template <bool FC_EPILOGUE>
__device__ __forceinline__ void gemm_tile(const __bf16* __restrict__ A,
                                          const __bf16* __restrict__ W,
                                          float* __restrict__ O,
                                          const float* __restrict__ bias,
                                          const float* __restrict__ resid,
                                          int m0, int n0) {
  __shared__ __attribute__((aligned(16))) __bf16 sX[2][128 * LDB];
  __shared__ __attribute__((aligned(16))) __bf16 sW[2][64 * LDB];
  const int tid = threadIdx.x;
  const int wid = tid >> 5;
  const int wm  = (wid & 3) * 32;
  const int wn  = (wid >> 2) * 32;
  const uint64_t xbase = (uint64_t)(uintptr_t)(A + (size_t)m0 * DMODEL);
  const uint64_t wbase = (uint64_t)(uintptr_t)(W + (size_t)n0 * DMODEL);

  auto issue = [&](int k0, int buf) {
    for (int i = tid; i < 1024; i += 256) {            // X: 128 rows x 8 chunks
      int r = i >> 3, c8 = i & 7;
      async_ld_b128(xbase, (uint32_t)((r * DMODEL + k0 + c8 * 8) * 2),
                    lds_off(&sX[buf][r * LDB + c8 * 8]));
    }
    for (int i = tid; i < 512; i += 256) {             // W: 64 rows x 8 chunks
      int r = i >> 3, c8 = i & 7;
      async_ld_b128(wbase, (uint32_t)((r * DMODEL + k0 + c8 * 8) * 2),
                    lds_off(&sW[buf][r * LDB + c8 * 8]));
    }
  };

  v8f c[2][2] = {};
  issue(0, 0);
  wait_async0();
  __syncthreads();
  int cur = 0;
  for (int k0 = 0; k0 < DMODEL; k0 += 64) {
    if (k0 + 64 < DMODEL) issue(k0 + 64, cur ^ 1);     // prefetch next tile
#pragma unroll
    for (int ks = 0; ks < 64; ks += 32) {
      v16bf a0 = a_frag_bf(sX[cur], LDB, wm, ks);
      v16bf a1 = a_frag_bf(sX[cur], LDB, wm + 16, ks);
      v16bf b0 = b_frag_nk_bf(sW[cur], LDB, wn, ks);
      v16bf b1 = b_frag_nk_bf(sW[cur], LDB, wn + 16, ks);
      c[0][0] = wmma_bf16(a0, b0, c[0][0]);
      c[0][1] = wmma_bf16(a0, b1, c[0][1]);
      c[1][0] = wmma_bf16(a1, b0, c[1][0]);
      c[1][1] = wmma_bf16(a1, b1, c[1][1]);
    }
    if (k0 + 64 < DMODEL) wait_async0();
    __syncthreads();
    cur ^= 1;
  }
  const int lane = tid & 31;
  const int colb = lane & 15;
  const int rbo  = (lane >> 4) * 8;
#pragma unroll
  for (int im = 0; im < 2; ++im)
#pragma unroll
    for (int in = 0; in < 2; ++in)
#pragma unroll
      for (int r = 0; r < 8; ++r) {
        int row = m0 + wm + im * 16 + rbo + r;
        int col = n0 + wn + in * 16 + colb;
        float v = c[im][in][r];
        if (FC_EPILOGUE) v += bias[col] + resid[(size_t)row * DMODEL + col];
        O[(size_t)row * DMODEL + col] = v;
      }
}

// ---------------- Kernel 1: fused QKV projection -----------------------------
__global__ __launch_bounds__(256)
void qkv_proj_kernel(const __bf16* __restrict__ X,
                     const __bf16* __restrict__ Wq, const __bf16* __restrict__ Wk,
                     const __bf16* __restrict__ Wv,
                     float* __restrict__ Q, float* __restrict__ K,
                     float* __restrict__ V) {
  const int m0  = blockIdx.x * 128;
  const int mat = blockIdx.y >> 3;
  const int n0  = (blockIdx.y & 7) * 64;
  const __bf16* W = (mat == 0) ? Wq : (mat == 1 ? Wk : Wv);
  float*        O = (mat == 0) ? Q  : (mat == 1 ? K  : V);
  gemm_tile<false>(X, W, O, nullptr, nullptr, m0, n0);
}

// ---------------- Kernel 3: FC + bias + residual -----------------------------
__global__ __launch_bounds__(256)
void fc_res_kernel(const __bf16* __restrict__ Xb, const __bf16* __restrict__ Wb,
                   const float* __restrict__ bias, const float* __restrict__ resid,
                   float* __restrict__ Y) {
  gemm_tile<true>(Xb, Wb, Y, bias, resid, blockIdx.x * 128, blockIdx.y * 64);
}

// ---------------- Kernel 2: star attention (stations + self loop) ------------
__global__ __launch_bounds__(256)
void star_attn_kernel(const float* __restrict__ Q, const float* __restrict__ K,
                      const float* __restrict__ V, const float* __restrict__ rpe,
                      __bf16* __restrict__ ctx) {
  const int LDSS = 132;   // f32 score pad
  __shared__ __attribute__((aligned(16))) __bf16 sQ[128 * LDB];
  __shared__ __attribute__((aligned(16))) __bf16 sK[128 * LDB];
  __shared__ __attribute__((aligned(16))) __bf16 sV[128 * LDB];
  __shared__ float  sS[128 * LDSS];
  __shared__ float  sSelf[128];
  __shared__ float  aSelf[128];
  const int i0 = blockIdx.x * 128;
  const int bh = blockIdx.y;                 // b*NHEAD + h
  const int b  = bh >> 3, h = bh & 7;
  const int tid = threadIdx.x;
  const int wid = tid >> 5;
  const int lane = tid & 31;

  // stage Q tile + station K/V as bf16 (f32 source -> single conversion)
  for (int i = tid; i < 128 * 64; i += 256) {
    int r = i >> 6, d = i & 63;
    sQ[r * LDB + d] = (__bf16)Q[((size_t)(b * NSEQ + i0 + r)) * DMODEL + h * DKH + d];
    sK[r * LDB + d] = (__bf16)K[((size_t)(b * NSEQ + r)) * DMODEL + h * DKH + d];
    sV[r * LDB + d] = (__bf16)V[((size_t)(b * NSEQ + r)) * DMODEL + h * DKH + d];
  }
  __syncthreads();

  // S = Q * K_st^T : 128x128, waves 4(M) x 2(N of 64)
  {
    const int wm = (wid & 3) * 32;
    const int wn = (wid >> 2) * 64;
    v8f c[2][4] = {};
#pragma unroll
    for (int ks = 0; ks < 64; ks += 32) {
      v16bf a0 = a_frag_bf(sQ, LDB, wm, ks);
      v16bf a1 = a_frag_bf(sQ, LDB, wm + 16, ks);
#pragma unroll
      for (int j = 0; j < 4; ++j) {
        v16bf bb = b_frag_nk_bf(sK, LDB, wn + j * 16, ks);
        c[0][j] = wmma_bf16(a0, bb, c[0][j]);
        c[1][j] = wmma_bf16(a1, bb, c[1][j]);
      }
    }
    const int colb = lane & 15;
    const int rbo  = (lane >> 4) * 8;
    const float* rpe_bh = rpe + ((size_t)bh * NSEQ + i0) * NSEQ;
#pragma unroll
    for (int im = 0; im < 2; ++im)
#pragma unroll
      for (int jn = 0; jn < 4; ++jn)
#pragma unroll
        for (int r = 0; r < 8; ++r) {
          int row = wm + im * 16 + rbo + r;
          int col = wn + jn * 16 + colb;
          sS[row * LDSS + col] =
              (c[im][jn][r] + rpe_bh[(size_t)row * NSEQ + col]) * 0.125f;
        }
  }
  // self-loop score (rows i >= NSTAT only; tiles are 128-aligned)
  if (tid < 128) {
    int r = tid, i = i0 + r;
    if (i >= NSTAT) {
      float dot = 0.f;
      const float* kg = &K[((size_t)(b * NSEQ + i)) * DMODEL + h * DKH];
#pragma unroll
      for (int d = 0; d < 64; ++d) dot += (float)sQ[r * LDB + d] * kg[d];
      sSelf[r] = (dot + rpe[(((size_t)bh * NSEQ) + i) * NSEQ + i]) * 0.125f;
    } else {
      sSelf[r] = -3.0e38f;    // diagonal already covered by stations
    }
  }
  __syncthreads();
  // row softmax over 128 stations (+ self)
  if (tid < 128) {
    int r = tid;
    float m = sSelf[r];
    for (int j = 0; j < 128; ++j) m = fmaxf(m, sS[r * LDSS + j]);
    float sum = 0.f;
    for (int j = 0; j < 128; ++j) {
      float e = __expf(sS[r * LDSS + j] - m);
      sS[r * LDSS + j] = e;
      sum += e;
    }
    float es = (sSelf[r] > -1.0e38f) ? __expf(sSelf[r] - m) : 0.f;
    sum += es;
    float inv = 1.f / sum;
    for (int j = 0; j < 128; ++j) sS[r * LDSS + j] *= inv;
    aSelf[r] = es * inv;
  }
  __syncthreads();
  // ctx = P * V_st (+ self term) : 128x64, waves 4(M) x 2(N of 32); bf16 out
  {
    const int wm = (wid & 3) * 32;
    const int wn = (wid >> 2) * 32;
    v8f c[2][2] = {};
#pragma unroll
    for (int j0 = 0; j0 < 128; j0 += 32) {
      v16bf a0 = a_frag_f32(sS, LDSS, wm, j0);
      v16bf a1 = a_frag_f32(sS, LDSS, wm + 16, j0);
      v16bf b0 = b_frag_kn_bf(sV, LDB, wn, j0);
      v16bf b1 = b_frag_kn_bf(sV, LDB, wn + 16, j0);
      c[0][0] = wmma_bf16(a0, b0, c[0][0]);
      c[0][1] = wmma_bf16(a0, b1, c[0][1]);
      c[1][0] = wmma_bf16(a1, b0, c[1][0]);
      c[1][1] = wmma_bf16(a1, b1, c[1][1]);
    }
    const int colb = lane & 15;
    const int rbo  = (lane >> 4) * 8;
#pragma unroll
    for (int im = 0; im < 2; ++im)
#pragma unroll
      for (int in = 0; in < 2; ++in)
#pragma unroll
        for (int r = 0; r < 8; ++r) {
          int row = wm + im * 16 + rbo + r;
          int d   = wn + in * 16 + colb;
          int i   = i0 + row;
          float val = c[im][in][r] +
                      aSelf[row] * V[((size_t)(b * NSEQ + i)) * DMODEL + h * DKH + d];
          ctx[((size_t)(b * NSEQ + i)) * DMODEL + h * DKH + d] = (__bf16)val;
        }
  }
}

// ---------------- Kernel 4: LayerNorm (row = 512) ----------------------------
__global__ __launch_bounds__(256)
void ln_kernel(const float* __restrict__ Y, const float* __restrict__ g,
               const float* __restrict__ bta, float* __restrict__ out) {
  const int row = blockIdx.x;
  const int t   = threadIdx.x;
  const float* yr = Y + (size_t)row * DMODEL;
  float x0 = yr[t], x1 = yr[t + 256];
  float s  = x0 + x1;
  float ss = x0 * x0 + x1 * x1;
#pragma unroll
  for (int o = 16; o > 0; o >>= 1) {        // wave32 reduction
    s  += __shfl_xor(s, o, 32);
    ss += __shfl_xor(ss, o, 32);
  }
  __shared__ float as[8], as2[8];
  if ((t & 31) == 0) { as[t >> 5] = s; as2[t >> 5] = ss; }
  __syncthreads();
  float tot = 0.f, tot2 = 0.f;
#pragma unroll
  for (int w = 0; w < 8; ++w) { tot += as[w]; tot2 += as2[w]; }
  float mu  = tot * (1.f / DMODEL);
  float var = tot2 * (1.f / DMODEL) - mu * mu;
  float inv = rsqrtf(var + 1e-6f);
  out[(size_t)row * DMODEL + t]       = (x0 - mu) * inv * g[t] + bta[t];
  out[(size_t)row * DMODEL + t + 256] = (x1 - mu) * inv * g[t + 256] + bta[t + 256];
}

// ---------------- launch -----------------------------------------------------
extern "C" void kernel_launch(void* const* d_in, const int* in_sizes, int n_in,
                              void* d_out, int out_size, void* d_ws, size_t ws_size,
                              hipStream_t stream) {
  const float* hs  = (const float*)d_in[0];
  const float* rpe = (const float*)d_in[1];
  const float* Wq  = (const float*)d_in[2];
  const float* Wk  = (const float*)d_in[3];
  const float* Wv  = (const float*)d_in[4];
  const float* fcw = (const float*)d_in[5];
  const float* fcb = (const float*)d_in[6];
  const float* lng = (const float*)d_in[7];
  const float* lnb = (const float*)d_in[8];
  float* out = (float*)d_out;

  const size_t S = (size_t)MROWS * DMODEL;      // 4,194,304
  const size_t WSZ = (size_t)DMODEL * DMODEL;   //   262,144
  float*  qw   = (float*)d_ws;
  float*  kw   = qw + S;
  float*  vw   = kw + S;
  float*  yw   = vw + S;
  __bf16* xb   = (__bf16*)(yw + S);
  __bf16* ctxb = xb + S;
  __bf16* wqb  = ctxb + S;
  __bf16* wkb  = wqb + WSZ;
  __bf16* wvb  = wkb + WSZ;
  __bf16* fwb  = wvb + WSZ;

  // one-shot bf16 casts (amortized over 24x / 64x reuse in the GEMMs)
  cvt_bf16_kernel<<<dim3((int)(S / 1024)), 256, 0, stream>>>(hs, xb, (int)S);
  cvt_bf16_kernel<<<dim3((int)(WSZ / 1024)), 256, 0, stream>>>(Wq, wqb, (int)WSZ);
  cvt_bf16_kernel<<<dim3((int)(WSZ / 1024)), 256, 0, stream>>>(Wk, wkb, (int)WSZ);
  cvt_bf16_kernel<<<dim3((int)(WSZ / 1024)), 256, 0, stream>>>(Wv, wvb, (int)WSZ);
  cvt_bf16_kernel<<<dim3((int)(WSZ / 1024)), 256, 0, stream>>>(fcw, fwb, (int)WSZ);

  qkv_proj_kernel<<<dim3(MROWS / 128, 24), 256, 0, stream>>>(xb, wqb, wkb, wvb,
                                                             qw, kw, vw);
  star_attn_kernel<<<dim3(NSEQ / 128, NBATCH * NHEAD), 256, 0, stream>>>(
      qw, kw, vw, rpe, ctxb);
  fc_res_kernel<<<dim3(MROWS / 128, DMODEL / 64), 256, 0, stream>>>(ctxb, fwb, fcb,
                                                                    hs, yw);
  ln_kernel<<<dim3(MROWS), 256, 0, stream>>>(yw, lng, lnb, out);
}
